// Model_40879498728890
// MI455X (gfx1250) — matrix operand order repaired and checked by reading, not compiled, not run on hardware
//
#include <hip/hip_runtime.h>
#include <hip/hip_bf16.h>
#include <math.h>

typedef __attribute__((ext_vector_type(16))) _Float16 v16h;
typedef __attribute__((ext_vector_type(8)))  float    v8f;
typedef __attribute__((ext_vector_type(4)))  int      v4i;

#define B_  8
#define H_  8
#define DM  512
#define DH  64
#define DFF 2048

#if __has_builtin(__builtin_amdgcn_global_load_async_to_lds_b128) && \
    __has_builtin(__builtin_amdgcn_s_wait_asynccnt)
#define HAVE_ASYNC_LDS 1
typedef __attribute__((address_space(1))) v4i* gptr_b128;
typedef __attribute__((address_space(3))) v4i* lptr_b128;
#else
#define HAVE_ASYNC_LDS 0
#endif

__device__ __forceinline__ float gelu_exact(float x) {
  return 0.5f * x * (1.0f + erff(x * 0.70710678118654752f));
}

// ---------------- weight conversion ----------------
__global__ void k_f32_to_f16(const float* __restrict__ s, _Float16* __restrict__ d, int n) {
  int i = blockIdx.x * blockDim.x + threadIdx.x;
  if (i < n) d[i] = (_Float16)s[i];
}

// cv_w: (2,512,512,3)[l][o][c][k] -> (2,512,1536)[l][o][k*512+c]
__global__ void k_cvw_rearrange(const float* __restrict__ cvw, _Float16* __restrict__ out) {
  int i = blockIdx.x * blockDim.x + threadIdx.x;
  const int total = 2 * 512 * 512 * 3;
  if (i >= total) return;
  int k = i % 3;
  int c = (i / 3) % 512;
  int o = (i / (3 * 512)) % 512;
  int l = i / (3 * 512 * 512);
  out[((size_t)l * 512 + o) * 1536 + k * 512 + c] = (_Float16)cvw[i];
}

// ---------------- token embedding + positional encoding ----------------
__global__ void k_tok_embed(const float* __restrict__ xe, const float* __restrict__ tw,
                            float* __restrict__ x, _Float16* __restrict__ xh, int L) {
  int i = blockIdx.x * blockDim.x + threadIdx.x;
  int total = B_ * L * DM;
  if (i >= total) return;
  int d = i % DM;
  int t = (i / DM) % L;
  int b = i / (DM * L);
  float acc = 0.f;
  #pragma unroll
  for (int k = 0; k < 3; ++k) {
    int tm = t + k - 1;
    tm = (tm < 0) ? tm + L : (tm >= L ? tm - L : tm);
    const float* xr = xe + ((size_t)b * L + tm) * 12;
    const float* wr = tw + d * 36 + k;     // tok_w[d][c][k], stride_c = 3
    #pragma unroll
    for (int c = 0; c < 12; ++c) acc += xr[c] * wr[c * 3];
  }
  int ii = d >> 1;
  float div = __expf(-(float)(2 * ii) * (9.210340371976184f / 512.0f));
  float ang = (float)t * div;
  acc += (d & 1) ? __cosf(ang) : __sinf(ang);
  x[i] = acc;
  xh[i] = (_Float16)acc;
}

// ---------------- WMMA GEMM: C[M,N] = A[M,K] * Bw[N,K]^T + bias (opt GELU) ----------------
// 128x128 macro-tile, 8 waves, wave tile 32x64 (2x4 WMMA 16x16x32 tiles).
// Double-buffered LDS; CDNA5 async global->LDS loads when available.
#define GT_M 128
#define GT_N 128
#define GT_K 32
#define LSTR 40   // LDS row stride in halves: 80B rows keep 16B chunk alignment,
                  // bank stride 20 -> conflict-free over 16 rows (gcd(20,64)=4)

__global__ void __launch_bounds__(256)
k_gemm_f16_wmma(const _Float16* __restrict__ A,   // M x K row-major (act, f16)
                const _Float16* __restrict__ Bw,  // N x K row-major (weight, f16)
                const float* __restrict__ bias,   // N
                float* __restrict__ C,            // M x N f32 (nullable)
                _Float16* __restrict__ Ch,        // M x N f16 (nullable)
                int M, int N, int K, int do_gelu) {
  __shared__ _Float16 sA[2][GT_M * LSTR];
  __shared__ _Float16 sB[2][GT_N * LSTR];

  const int tid     = threadIdx.x;
  const int lane    = tid & 31;
  const int wave    = tid >> 5;
  const int halfSel = lane >> 4;
  const int l16     = lane & 15;

  const int blockM = blockIdx.y * GT_M;
  const int blockN = blockIdx.x * GT_N;
  const int waveM  = (wave & 3) * 32;   // 4 waves in M
  const int waveN  = (wave >> 2) * 64;  // 2 waves in N

  // per-thread tile-load coordinates: 128 rows x 32 halves = 512 x 16B chunks,
  // 256 threads -> 2 chunks each (rows r and r+64)
  const int rowL = tid >> 2;
  const int kofL = (tid & 3) * 8;

  const v8f vzero = {0.f, 0.f, 0.f, 0.f, 0.f, 0.f, 0.f, 0.f};
  v8f acc[2][4];
  #pragma unroll
  for (int i = 0; i < 2; ++i)
    #pragma unroll
    for (int j = 0; j < 4; ++j) acc[i][j] = vzero;

  // row clamp for A (padding rows feed only masked outputs)
  const int rA0 = min(blockM + rowL,      M - 1);
  const int rA1 = min(blockM + rowL + 64, M - 1);
  const int rB0 = blockN + rowL;        // N is a multiple of 128
  const int rB1 = blockN + rowL + 64;

  auto issue = [&](int k0, int buf) {
    const _Float16* gA0 = A  + (size_t)rA0 * K + k0 + kofL;
    const _Float16* gA1 = A  + (size_t)rA1 * K + k0 + kofL;
    const _Float16* gB0 = Bw + (size_t)rB0 * K + k0 + kofL;
    const _Float16* gB1 = Bw + (size_t)rB1 * K + k0 + kofL;
    _Float16* lA0 = &sA[buf][rowL * LSTR + kofL];
    _Float16* lA1 = &sA[buf][(rowL + 64) * LSTR + kofL];
    _Float16* lB0 = &sB[buf][rowL * LSTR + kofL];
    _Float16* lB1 = &sB[buf][(rowL + 64) * LSTR + kofL];
#if HAVE_ASYNC_LDS
    __builtin_amdgcn_global_load_async_to_lds_b128((gptr_b128)gA0, (lptr_b128)lA0, 0, 0);
    __builtin_amdgcn_global_load_async_to_lds_b128((gptr_b128)gA1, (lptr_b128)lA1, 0, 0);
    __builtin_amdgcn_global_load_async_to_lds_b128((gptr_b128)gB0, (lptr_b128)lB0, 0, 0);
    __builtin_amdgcn_global_load_async_to_lds_b128((gptr_b128)gB1, (lptr_b128)lB1, 0, 0);
#else
    uint4 a0 = *(const uint4*)gA0;
    uint4 a1 = *(const uint4*)gA1;
    uint4 b0 = *(const uint4*)gB0;
    uint4 b1 = *(const uint4*)gB1;
    *(uint4*)lA0 = a0;
    *(uint4*)lA1 = a1;
    *(uint4*)lB0 = b0;
    *(uint4*)lB1 = b1;
    if (k0 + GT_K < K) {
      __builtin_prefetch((const void*)(gA0 + GT_K), 0, 1);
      __builtin_prefetch((const void*)(gB0 + GT_K), 0, 1);
    }
#endif
  };

  // prologue: fill buffer 0
  issue(0, 0);
#if HAVE_ASYNC_LDS
  __builtin_amdgcn_s_wait_asynccnt(0);
#endif
  __syncthreads();

  int buf = 0;
  for (int k0 = 0; k0 < K; k0 += GT_K) {
    const int nbuf = buf ^ 1;
    const bool more = (k0 + GT_K) < K;
    if (more) issue(k0 + GT_K, nbuf);   // overlap with compute below

    union Frag { v16h v; unsigned int u[8]; };
    Frag a[2], b[4];
    const _Float16* bufA = sA[buf];
    const _Float16* bufB = sB[buf];
    #pragma unroll
    for (int mt = 0; mt < 2; ++mt) {
      const _Float16* base = bufA + (waveM + mt * 16 + l16) * LSTR;
      #pragma unroll
      for (int r = 0; r < 8; ++r) {
        int kk = ((r & 4) ? 16 : 0) + halfSel * 8 + (r & 3) * 2;
        a[mt].u[r] = *(const unsigned int*)(base + kk);
      }
    }
    #pragma unroll
    for (int nt = 0; nt < 4; ++nt) {
      const _Float16* base = bufB + (waveN + nt * 16 + l16) * LSTR;
      #pragma unroll
      for (int r = 0; r < 8; ++r) {
        int kk = ((r & 4) ? 16 : 0) + halfSel * 8 + (r & 3) * 2;
        b[nt].u[r] = *(const unsigned int*)(base + kk);
      }
    }
    #pragma unroll
    for (int mt = 0; mt < 2; ++mt)
      #pragma unroll
      for (int nt = 0; nt < 4; ++nt)
        acc[mt][nt] = __builtin_amdgcn_wmma_f32_16x16x32_f16(
            false, a[mt].v, false, b[nt].v, (short)0, acc[mt][nt], false, false);

    if (more) {
#if HAVE_ASYNC_LDS
      __builtin_amdgcn_s_wait_asynccnt(0);
#endif
      __syncthreads();
    }
    buf = nbuf;
  }

  #pragma unroll
  for (int mt = 0; mt < 2; ++mt) {
    #pragma unroll
    for (int nt = 0; nt < 4; ++nt) {
      #pragma unroll
      for (int r = 0; r < 8; ++r) {
        int m = blockM + waveM + mt * 16 + halfSel * 8 + r;
        int n = blockN + waveN + nt * 16 + l16;
        if (m < M) {
          float v = acc[mt][nt][r] + bias[n];
          if (do_gelu) v = gelu_exact(v);
          size_t o = (size_t)m * N + n;
          if (C)  C[o]  = v;
          if (Ch) Ch[o] = (_Float16)v;
        }
      }
    }
  }
}

// ---------------- ProbSparse attention pieces ----------------
__global__ void k_attn_measure(const float* __restrict__ q, const float* __restrict__ k,
                               float* __restrict__ mm, int L, int nS) {
  int i = blockIdx.x * blockDim.x + threadIdx.x;
  int total = B_ * H_ * L;
  if (i >= total) return;
  int l  = i % L;
  int bh = i / L;
  int h  = bh % H_;
  int b  = bh / H_;
  const float* qr = q + ((size_t)b * L + l) * DM + h * DH;
  unsigned int st = (unsigned int)l * 2654435761u + 12345u;
  float mx = -1e30f, sm = 0.f;
  for (int s = 0; s < nS; ++s) {
    st = st * 1664525u + 1013904223u;
    int ls = (int)((st >> 8) % (unsigned int)L);
    const float* kr = k + ((size_t)b * L + ls) * DM + h * DH;
    float d = 0.f;
    #pragma unroll 8
    for (int j = 0; j < DH; ++j) d += qr[j] * kr[j];
    mx = fmaxf(mx, d);
    sm += d;
  }
  mm[i] = mx - sm / (float)L;
}

__global__ void __launch_bounds__(256)
k_attn_topk(const float* __restrict__ mm, int* __restrict__ topidx, int L, int nTop) {
  __shared__ float vals[1024];
  __shared__ float rv[256];
  __shared__ int   ri[256];
  int bh = blockIdx.x;
  const float* src = mm + (size_t)bh * L;
  for (int i = threadIdx.x; i < 1024; i += 256) vals[i] = (i < L) ? src[i] : -1e30f;
  __syncthreads();
  for (int t = 0; t < nTop; ++t) {
    float best = -1e30f; int bi = 0;
    for (int i = threadIdx.x; i < L; i += 256)
      if (vals[i] > best) { best = vals[i]; bi = i; }
    rv[threadIdx.x] = best; ri[threadIdx.x] = bi;
    __syncthreads();
    for (int s = 128; s > 0; s >>= 1) {
      if (threadIdx.x < s && rv[threadIdx.x + s] > rv[threadIdx.x]) {
        rv[threadIdx.x] = rv[threadIdx.x + s];
        ri[threadIdx.x] = ri[threadIdx.x + s];
      }
      __syncthreads();
    }
    if (threadIdx.x == 0) { topidx[(size_t)bh * nTop + t] = ri[0]; vals[ri[0]] = -1e30f; }
    __syncthreads();
  }
}

__global__ void k_vmean(const float* __restrict__ v, float* __restrict__ vm, int L) {
  int i = blockIdx.x * blockDim.x + threadIdx.x;
  if (i >= B_ * H_ * DH) return;
  int d = i % DH;
  int h = (i / DH) % H_;
  int b = i / (DH * H_);
  float s = 0.f;
  for (int t = 0; t < L; ++t) s += v[((size_t)b * L + t) * DM + h * DH + d];
  vm[i] = s / (float)L;
}

__global__ void k_ctx_fill(const float* __restrict__ vm, float* __restrict__ ctx,
                           _Float16* __restrict__ ctxh, int L) {
  int i = blockIdx.x * blockDim.x + threadIdx.x;
  int total = B_ * L * DM;
  if (i >= total) return;
  int c = i % DM;
  int b = i / (DM * L);
  float val = vm[(b * H_ + (c / DH)) * DH + (c % DH)];
  ctx[i] = val;
  ctxh[i] = (_Float16)val;
}

__global__ void __launch_bounds__(256)
k_attn_topq(const float* __restrict__ q, const float* __restrict__ k,
            const float* __restrict__ v, const int* __restrict__ topidx,
            float* __restrict__ ctx, _Float16* __restrict__ ctxh, int L, int nTop) {
  __shared__ float qs[DH];
  __shared__ float sc[1024];
  __shared__ float red[256];
  int u  = blockIdx.x;
  int bh = blockIdx.y;
  int h = bh % H_, b = bh / H_;
  int l0 = topidx[(size_t)bh * nTop + u];
  const size_t rowQ = ((size_t)b * L + l0) * DM + h * DH;
  if (threadIdx.x < DH) qs[threadIdx.x] = q[rowQ + threadIdx.x];
  __syncthreads();
  float lmax = -1e30f;
  for (int l = threadIdx.x; l < L; l += 256) {
    const float* kr = k + ((size_t)b * L + l) * DM + h * DH;
    float d = 0.f;
    #pragma unroll 8
    for (int j = 0; j < DH; ++j) d += qs[j] * kr[j];
    d *= 0.125f;                      // 1/sqrt(64)
    sc[l] = d;
    lmax = fmaxf(lmax, d);
  }
  red[threadIdx.x] = lmax; __syncthreads();
  for (int s = 128; s > 0; s >>= 1) {
    if (threadIdx.x < s) red[threadIdx.x] = fmaxf(red[threadIdx.x], red[threadIdx.x + s]);
    __syncthreads();
  }
  float smax = red[0]; __syncthreads();
  float lsum = 0.f;
  for (int l = threadIdx.x; l < L; l += 256) {
    float e = __expf(sc[l] - smax);
    sc[l] = e;
    lsum += e;
  }
  red[threadIdx.x] = lsum; __syncthreads();
  for (int s = 128; s > 0; s >>= 1) {
    if (threadIdx.x < s) red[threadIdx.x] += red[threadIdx.x + s];
    __syncthreads();
  }
  float inv = 1.f / red[0];
  __syncthreads();
  if (threadIdx.x < DH) {
    int d = threadIdx.x;
    float a = 0.f;
    for (int l = 0; l < L; ++l) a += sc[l] * v[((size_t)b * L + l) * DM + h * DH + d];
    a *= inv;
    ctx[rowQ + d] = a;
    ctxh[rowQ + d] = (_Float16)a;
  }
}

// ---------------- residual + layernorm ----------------
__global__ void __launch_bounds__(256)
k_add_ln(const float* __restrict__ x, const float* __restrict__ y,
         const float* __restrict__ w, const float* __restrict__ bb,
         float* __restrict__ out, _Float16* __restrict__ outh) {
  __shared__ float red[256];
  size_t row = blockIdx.x;
  const float* xr = x + row * DM;
  const float* yr = y + row * DM;
  int t = threadIdx.x;
  float v0 = xr[t] + yr[t];
  float v1 = xr[t + 256] + yr[t + 256];
  red[t] = v0 + v1; __syncthreads();
  for (int s = 128; s > 0; s >>= 1) { if (t < s) red[t] += red[t + s]; __syncthreads(); }
  float mu = red[0] * (1.0f / DM); __syncthreads();
  float d0 = v0 - mu, d1 = v1 - mu;
  red[t] = d0 * d0 + d1 * d1; __syncthreads();
  for (int s = 128; s > 0; s >>= 1) { if (t < s) red[t] += red[t + s]; __syncthreads(); }
  float rs = rsqrtf(red[0] * (1.0f / DM) + 1e-5f);
  float o0 = d0 * rs * w[t] + bb[t];
  float o1 = d1 * rs * w[t + 256] + bb[t + 256];
  out[row * DM + t] = o0;       out[row * DM + t + 256] = o1;
  outh[row * DM + t] = (_Float16)o0; outh[row * DM + t + 256] = (_Float16)o1;
}

// ---------------- distilling conv: im2row (circular) ----------------
__global__ void k_im2row(const float* __restrict__ x, _Float16* __restrict__ r, int L) {
  size_t i = (size_t)blockIdx.x * blockDim.x + threadIdx.x;
  size_t total = (size_t)B_ * L * 1536;
  if (i >= total) return;
  int c = (int)(i % 1536);
  int t = (int)((i / 1536) % L);
  int b = (int)(i / ((size_t)1536 * L));
  int kk = c / 512, cc = c % 512;
  int tm = t + kk - 1;
  tm = tm < 0 ? tm + L : (tm >= L ? tm - L : tm);
  r[i] = (_Float16)x[((size_t)b * L + tm) * DM + cc];
}

__global__ void k_bn_elu_pool(const float* __restrict__ z, const float* __restrict__ bw,
                              const float* __restrict__ bb, float* __restrict__ xo,
                              _Float16* __restrict__ xho, int Lin, int Lout) {
  int i = blockIdx.x * blockDim.x + threadIdx.x;
  int total = B_ * Lout * DM;
  if (i >= total) return;
  int d  = i % DM;
  int tp = (i / DM) % Lout;
  int b  = i / (DM * Lout);
  float sc = bw[d] * rsqrtf(1.0f + 1e-5f);
  float sh = bb[d];
  float best = -1e30f;
  #pragma unroll
  for (int dt = -1; dt <= 1; ++dt) {
    int t = 2 * tp + dt;
    if (t < 0 || t >= Lin) continue;
    float val = z[((size_t)b * Lin + t) * DM + d] * sc + sh;
    val = val > 0.f ? val : (__expf(val) - 1.f);   // ELU
    best = fmaxf(best, val);
  }
  xo[i] = best;
  xho[i] = (_Float16)best;
}

// ---------------- final LN + GELU ----------------
__global__ void __launch_bounds__(256)
k_ln_gelu(const float* __restrict__ x, const float* __restrict__ w,
          const float* __restrict__ bb, float* __restrict__ out) {
  __shared__ float red[256];
  size_t row = blockIdx.x;
  const float* xr = x + row * DM;
  int t = threadIdx.x;
  float v0 = xr[t], v1 = xr[t + 256];
  red[t] = v0 + v1; __syncthreads();
  for (int s = 128; s > 0; s >>= 1) { if (t < s) red[t] += red[t + s]; __syncthreads(); }
  float mu = red[0] * (1.0f / DM); __syncthreads();
  float d0 = v0 - mu, d1 = v1 - mu;
  red[t] = d0 * d0 + d1 * d1; __syncthreads();
  for (int s = 128; s > 0; s >>= 1) { if (t < s) red[t] += red[t + s]; __syncthreads(); }
  float rs = rsqrtf(red[0] * (1.0f / DM) + 1e-5f);
  out[row * DM + t]       = gelu_exact(d0 * rs * w[t] + bb[t]);
  out[row * DM + t + 256] = gelu_exact(d1 * rs * w[t + 256] + bb[t + 256]);
}

// ---------------- final projection (8 x 131072) @ (10 x 131072)^T ----------------
__global__ void __launch_bounds__(256)
k_proj(const float* __restrict__ g, const float* __restrict__ pw,
       const float* __restrict__ pb, float* __restrict__ out) {
  __shared__ float red[256];
  int cls = blockIdx.x, b = blockIdx.y;
  const float* gr = g + (size_t)b * (DM * 256);
  const float* wr = pw + (size_t)cls * (DM * 256);
  float s = 0.f;
  for (int i = threadIdx.x; i < DM * 256; i += 256) s += gr[i] * wr[i];
  red[threadIdx.x] = s; __syncthreads();
  for (int st = 128; st > 0; st >>= 1) {
    if (threadIdx.x < st) red[threadIdx.x] += red[threadIdx.x + st];
    __syncthreads();
  }
  if (threadIdx.x == 0) out[b * 10 + cls] = red[0] + pb[cls];
}

// =====================================================================
extern "C" void kernel_launch(void* const* d_in, const int* in_sizes, int n_in,
                              void* d_out, int out_size, void* d_ws, size_t ws_size,
                              hipStream_t stream) {
  (void)in_sizes; (void)n_in; (void)out_size; (void)ws_size;
  const float* x_enc   = (const float*)d_in[0];
  const float* tok_w   = (const float*)d_in[1];
  const float* attn_wq = (const float*)d_in[2];
  const float* attn_bq = (const float*)d_in[3];
  const float* attn_wk = (const float*)d_in[4];
  const float* attn_bk = (const float*)d_in[5];
  const float* attn_wv = (const float*)d_in[6];
  const float* attn_bv = (const float*)d_in[7];
  const float* attn_wo = (const float*)d_in[8];
  const float* attn_bo = (const float*)d_in[9];
  const float* ln1_w   = (const float*)d_in[10];
  const float* ln1_b   = (const float*)d_in[11];
  const float* ln2_w   = (const float*)d_in[12];
  const float* ln2_b   = (const float*)d_in[13];
  const float* ff1_w   = (const float*)d_in[14];
  const float* ff1_b   = (const float*)d_in[15];
  const float* ff2_w   = (const float*)d_in[16];
  const float* ff2_b   = (const float*)d_in[17];
  const float* cv_w    = (const float*)d_in[18];
  const float* cv_b    = (const float*)d_in[19];
  const float* bn_w    = (const float*)d_in[20];
  const float* bn_b    = (const float*)d_in[21];
  const float* norm_w  = (const float*)d_in[22];
  const float* norm_b  = (const float*)d_in[23];
  const float* proj_w  = (const float*)d_in[24];
  const float* proj_b  = (const float*)d_in[25];

  // ---- workspace carve ----
  char* base = (char*)d_ws;
  size_t off = 0;
  auto alloc = [&](size_t bytes) -> void* {
    void* p = base + off;
    off += (bytes + 255) & ~(size_t)255;
    return p;
  };
  const int L0 = 1021, M0 = B_ * L0;
  float* x   = (float*)alloc((size_t)M0 * DM * 4);
  float* y   = (float*)alloc((size_t)M0 * DM * 4);
  float* q   = (float*)alloc((size_t)M0 * DM * 4);
  float* kk  = (float*)alloc((size_t)M0 * DM * 4);
  float* v   = (float*)alloc((size_t)M0 * DM * 4);
  float* cz  = (float*)alloc((size_t)M0 * DM * 4);   // ctx f32 / conv out
  float* mm  = (float*)alloc((size_t)B_ * H_ * L0 * 4);
  float* vm  = (float*)alloc((size_t)B_ * H_ * DH * 4);
  int*   topidx = (int*)alloc((size_t)B_ * H_ * 64 * 4);
  _Float16* xh  = (_Float16*)alloc((size_t)M0 * DM * 2);
  _Float16* ch  = (_Float16*)alloc((size_t)M0 * DM * 2);
  _Float16* f1h = (_Float16*)alloc((size_t)M0 * DFF * 2);  // also reused as im2row buf
  _Float16* wqh = (_Float16*)alloc((size_t)3 * DM * DM * 2);
  _Float16* wkh = (_Float16*)alloc((size_t)3 * DM * DM * 2);
  _Float16* wvh = (_Float16*)alloc((size_t)3 * DM * DM * 2);
  _Float16* woh = (_Float16*)alloc((size_t)3 * DM * DM * 2);
  _Float16* f1w = (_Float16*)alloc((size_t)3 * DFF * DM * 2);
  _Float16* f2w = (_Float16*)alloc((size_t)3 * DM * DFF * 2);
  _Float16* cvwh = (_Float16*)alloc((size_t)2 * DM * 1536 * 2);
  _Float16* r16 = f1h;   // im2row (M*1536 <= M*2048)
  float* gx = y;         // final gelu output reuses y (2048*512 <= M0*512)

  // ---- weight conversion ----
  auto cvt = [&](const float* s, _Float16* d, size_t n) {
    k_f32_to_f16<<<(unsigned)((n + 255) / 256), 256, 0, stream>>>(s, d, (int)n);
  };
  cvt(attn_wq, wqh, (size_t)3 * DM * DM);
  cvt(attn_wk, wkh, (size_t)3 * DM * DM);
  cvt(attn_wv, wvh, (size_t)3 * DM * DM);
  cvt(attn_wo, woh, (size_t)3 * DM * DM);
  cvt(ff1_w,   f1w, (size_t)3 * DFF * DM);
  cvt(ff2_w,   f2w, (size_t)3 * DM * DFF);
  k_cvw_rearrange<<<(2 * 512 * 512 * 3 + 255) / 256, 256, 0, stream>>>(cv_w, cvwh);

  // ---- token embedding + PE ----
  k_tok_embed<<<(M0 * DM + 255) / 256, 256, 0, stream>>>(x_enc, tok_w, x, xh, L0);

  auto gemm = [&](const _Float16* A, const _Float16* Bw, const float* bias,
                  float* C, _Float16* Chp, int M, int N, int K, int do_gelu) {
    dim3 g(N / GT_N, (M + GT_M - 1) / GT_M);
    k_gemm_f16_wmma<<<g, 256, 0, stream>>>(A, Bw, bias, C, Chp, M, N, K, do_gelu);
  };

  const int Ls[3]    = {1021, 511, 256};
  const int nTops[3] = {35, 35, 30};   // min(5*ceil(ln L), L)
  for (int l = 0; l < 3; ++l) {
    const int L = Ls[l], M = B_ * L, nTop = nTops[l];
    // QKV projections
    gemm(xh, wqh + (size_t)l * DM * DM, attn_bq + l * DM, q,  nullptr, M, DM, DM, 0);
    gemm(xh, wkh + (size_t)l * DM * DM, attn_bk + l * DM, kk, nullptr, M, DM, DM, 0);
    gemm(xh, wvh + (size_t)l * DM * DM, attn_bv + l * DM, v,  nullptr, M, DM, DM, 0);
    // ProbSparse attention
    k_attn_measure<<<(B_ * H_ * L + 255) / 256, 256, 0, stream>>>(q, kk, mm, L, nTop);
    k_attn_topk<<<B_ * H_, 256, 0, stream>>>(mm, topidx, L, nTop);
    k_vmean<<<(B_ * H_ * DH + 255) / 256, 256, 0, stream>>>(v, vm, L);
    k_ctx_fill<<<(M * DM + 255) / 256, 256, 0, stream>>>(vm, cz, ch, L);
    dim3 gq(nTop, B_ * H_);
    k_attn_topq<<<gq, 256, 0, stream>>>(q, kk, v, topidx, cz, ch, L, nTop);
    // output projection + residual LN
    gemm(ch, woh + (size_t)l * DM * DM, attn_bo + l * DM, y, nullptr, M, DM, DM, 0);
    k_add_ln<<<M, 256, 0, stream>>>(x, y, ln1_w + l * DM, ln1_b + l * DM, x, xh);
    // FFN
    gemm(xh,  f1w + (size_t)l * DFF * DM, ff1_b + l * DFF, nullptr, f1h, M, DFF, DM, 1);
    gemm(f1h, f2w + (size_t)l * DM * DFF, ff2_b + l * DM,  y, nullptr, M, DM, DFF, 0);
    k_add_ln<<<M, 256, 0, stream>>>(x, y, ln2_w + l * DM, ln2_b + l * DM, x, xh);
    // distilling conv layer
    if (l < 2) {
      const int Lout = (L - 1) / 2 + 1;
      k_im2row<<<(int)(((size_t)M * 1536 + 255) / 256), 256, 0, stream>>>(x, r16, L);
      gemm(r16, cvwh + (size_t)l * DM * 1536, cv_b + l * DM, cz, nullptr, M, DM, 1536, 0);
      k_bn_elu_pool<<<(B_ * Lout * DM + 255) / 256, 256, 0, stream>>>(
          cz, bn_w + l * DM, bn_b + l * DM, x, xh, L, Lout);
    }
  }
  // final norm + gelu + projection
  k_ln_gelu<<<B_ * 256, 256, 0, stream>>>(x, norm_w, norm_b, gx);
  dim3 gp(10, B_);
  k_proj<<<gp, 256, 0, stream>>>(gx, proj_w, proj_b, (float*)d_out);
}
